// AttentionLayer_start_88081189306584
// MI455X (gfx1250) — matrix operand order repaired and checked by reading, not compiled, not run on hardware
//
#include <hip/hip_runtime.h>
#include <hip/hip_bf16.h>

// MI455X / gfx1250 implementation.
//  Kernel 0: one-shot W fp32 -> bf16 conversion into workspace (8 MiB).
//  Kernel 1: fused per-(b,h) attention with v_wmma_f32_16x16x32_bf16;
//            X = context reshaped to [8192,2048] bf16 into workspace (32 MiB).
//  Kernel 2: projection GEMM out = X @ W^T + b, bf16 WMMA, f32 accumulate,
//            double-buffered LDS pipeline (1 barrier / 32-wide K chunk).

typedef __attribute__((ext_vector_type(16))) __bf16 v16bf;
typedef __attribute__((ext_vector_type(8)))  __bf16 v8bf;
typedef __attribute__((ext_vector_type(2)))  __bf16 v2bf;
typedef __attribute__((ext_vector_type(8)))  float  v8f;

// Native clang __bf16 conversion (RNE); trivially copyable for bit views.
static __device__ __forceinline__ unsigned short f2bf(float f) {
  __bf16 h = (__bf16)f;
  return __builtin_bit_cast(unsigned short, h);
}

static __device__ __forceinline__ unsigned int pk2bf(float a, float b) {
  v2bf p;
  p[0] = (__bf16)a;
  p[1] = (__bf16)b;
  return __builtin_bit_cast(unsigned int, p);
}

static __device__ __forceinline__ v16bf combine16(v8bf lo, v8bf hi) {
  return __builtin_shufflevector(lo, hi, 0, 1, 2, 3, 4, 5, 6, 7,
                                 8, 9, 10, 11, 12, 13, 14, 15);
}

// ---------------------------------------------------------------------------
// Kernel 0: W fp32 -> bf16 (2048x2048).  2048 blocks x 256 thr x 8 elems.
// ---------------------------------------------------------------------------
__launch_bounds__(256)
__global__ void wcvt_kernel(const float* __restrict__ w,
                            unsigned short* __restrict__ wbf) {
  const long long i = ((long long)blockIdx.x * 256 + threadIdx.x) * 8;
  float4 f0 = *(const float4*)&w[i];
  float4 f1 = *(const float4*)&w[i + 4];
  uint4 o;
  o.x = pk2bf(f0.x, f0.y);
  o.y = pk2bf(f0.z, f0.w);
  o.z = pk2bf(f1.x, f1.y);
  o.w = pk2bf(f1.z, f1.w);
  *(uint4*)&wbf[i] = o;
}

// ---------------------------------------------------------------------------
// Kernel 1: attention per (b,h).  grid = 2048 blocks, block = 128 (4 waves).
// ---------------------------------------------------------------------------
__launch_bounds__(128)
__global__ void attn_bf16_wmma_kernel(const float* __restrict__ q,
                                      const float* __restrict__ k,
                                      const float* __restrict__ v,
                                      unsigned short* __restrict__ xws) {
  __shared__ __align__(32) unsigned short lds_q[64 * 128];   // Q row-major
  __shared__ __align__(32) unsigned short lds_k[64 * 128];   // K row-major
  __shared__ __align__(32) unsigned short lds_vt[128 * 64];  // V transposed [d][s]
  __shared__ __align__(32) unsigned short lds_a[64 * 64];    // softmax(A), wave-private rows

  const int tid  = threadIdx.x;
  const int lane = tid & 31;
  const int wave = tid >> 5;       // 0..3
  const int g    = (lane >> 4) & 1;
  const int ln   = lane & 15;

  const long long base = (long long)blockIdx.x * (64 * 128);
  const float4* qg = (const float4*)(q + base);
  const float4* kg = (const float4*)(k + base);
  const float4* vg = (const float4*)(v + base);

  // Stage Q,K (row major) and V (transposed) into LDS as bf16.
#pragma unroll
  for (int i = 0; i < 16; ++i) {
    int fi  = tid + i * 128;        // float4 index, 2048 total
    int row = fi >> 5;              // 32 float4 per 128-elem row
    int c4  = fi & 31;
    float4 fq = qg[fi];
    float4 fk = kg[fi];
    float4 fv = vg[fi];
    uint2 sq = {pk2bf(fq.x, fq.y), pk2bf(fq.z, fq.w)};
    uint2 sk = {pk2bf(fk.x, fk.y), pk2bf(fk.z, fk.w)};
    *(uint2*)&lds_q[row * 128 + c4 * 4] = sq;
    *(uint2*)&lds_k[row * 128 + c4 * 4] = sk;
    int d0 = c4 * 4;
    lds_vt[(d0 + 0) * 64 + row] = f2bf(fv.x);
    lds_vt[(d0 + 1) * 64 + row] = f2bf(fv.y);
    lds_vt[(d0 + 2) * 64 + row] = f2bf(fv.z);
    lds_vt[(d0 + 3) * 64 + row] = f2bf(fv.w);
  }
  __syncthreads();

  const int m0 = wave * 16;        // this wave's 16 query rows

  // ---- scores S = Q * K^T : 4 col-tiles x 4 K-chunks of 32 ----
  v8f sacc[4];
  const v8f vzero = {};
#pragma unroll
  for (int t = 0; t < 4; ++t) sacc[t] = vzero;

#pragma unroll
  for (int kc = 0; kc < 4; ++kc) {
    const int k0 = kc * 32;
    const int arow = m0 + ln;
    v8bf a0 = *(const v8bf*)&lds_q[arow * 128 + k0 + g * 16];
    v8bf a1 = *(const v8bf*)&lds_q[arow * 128 + k0 + g * 16 + 8];
    v16bf af = combine16(a0, a1);
#pragma unroll
    for (int t = 0; t < 4; ++t) {
      const int col = t * 16 + ln;
      v16bf bf = *(const v16bf*)&lds_k[col * 128 + k0 + g * 16];
      sacc[t] = __builtin_amdgcn_wmma_f32_16x16x32_bf16(
          false, af, false, bf, (short)0, sacc[t], false, false);
    }
  }

  // ---- softmax(0.1 * S) row-wise; write bf16 A to wave-private LDS ----
#pragma unroll
  for (int r = 0; r < 8; ++r) {
    float t0 = 0.1f * sacc[0][r];
    float t1 = 0.1f * sacc[1][r];
    float t2 = 0.1f * sacc[2][r];
    float t3 = 0.1f * sacc[3][r];
    float mx = fmaxf(fmaxf(t0, t1), fmaxf(t2, t3));
#pragma unroll
    for (int m = 1; m <= 8; m <<= 1) mx = fmaxf(mx, __shfl_xor(mx, m, 32));
    float e0 = __expf(t0 - mx);
    float e1 = __expf(t1 - mx);
    float e2 = __expf(t2 - mx);
    float e3 = __expf(t3 - mx);
    float sm = (e0 + e1) + (e2 + e3);
#pragma unroll
    for (int m = 1; m <= 8; m <<= 1) sm += __shfl_xor(sm, m, 32);
    float inv = 1.0f / sm;
    const int row = m0 + g * 8 + r;
    lds_a[row * 64 + 0  + ln] = f2bf(e0 * inv);
    lds_a[row * 64 + 16 + ln] = f2bf(e1 * inv);
    lds_a[row * 64 + 32 + ln] = f2bf(e2 * inv);
    lds_a[row * 64 + 48 + ln] = f2bf(e3 * inv);
  }
  // lds_a rows [m0,m0+16) are produced and consumed by this wave only:
  // same-wave LDS ordering makes a barrier unnecessary.

  // ---- O = A * V : 8 col-tiles of 16 over D=128, K=64 in 2 chunks ----
  v8f oacc[8];
#pragma unroll
  for (int t = 0; t < 8; ++t) oacc[t] = vzero;

#pragma unroll
  for (int kc = 0; kc < 2; ++kc) {
    const int k0 = kc * 32;
    const int arow = m0 + ln;
    v8bf a0 = *(const v8bf*)&lds_a[arow * 64 + k0 + g * 16];
    v8bf a1 = *(const v8bf*)&lds_a[arow * 64 + k0 + g * 16 + 8];
    v16bf af = combine16(a0, a1);
#pragma unroll
    for (int t = 0; t < 8; ++t) {
      const int d = t * 16 + ln;
      v16bf bf = *(const v16bf*)&lds_vt[d * 64 + k0 + g * 16];
      oacc[t] = __builtin_amdgcn_wmma_f32_16x16x32_bf16(
          false, af, false, bf, (short)0, oacc[t], false, false);
    }
  }

  // ---- store X[b*64 + l][h*128 + d] as bf16 ----
  const int bb = blockIdx.x >> 4;
  const int hh = blockIdx.x & 15;
#pragma unroll
  for (int t = 0; t < 8; ++t) {
    const int d = t * 16 + ln;
#pragma unroll
    for (int r = 0; r < 8; ++r) {
      const int l = m0 + g * 8 + r;
      xws[(long long)(bb * 64 + l) * 2048 + hh * 128 + d] = f2bf(oacc[t][r]);
    }
  }
}

// ---------------------------------------------------------------------------
// Kernel 2: out[8192x2048] = X(bf16) @ Wbf^T + b.
// grid = (N/128=16, M/128=64), block = 256 (8 waves, 2x4 wave grid).
// Double-buffered LDS, one barrier per 32-wide K chunk.
// ---------------------------------------------------------------------------
__launch_bounds__(256)
__global__ void proj_bf16_wmma_kernel(const unsigned short* __restrict__ x,
                                      const unsigned short* __restrict__ wbf,
                                      const float* __restrict__ bias,
                                      float* __restrict__ out) {
  __shared__ __align__(32) unsigned short lds_x[2][128 * 32];  // [m][k] bf16
  __shared__ __align__(32) unsigned short lds_w[2][128 * 32];  // [n][k] bf16

  const int tid  = threadIdx.x;
  const int lane = tid & 31;
  const int wave = tid >> 5;
  const int g    = (lane >> 4) & 1;
  const int ln   = lane & 15;
  const int wm   = wave >> 2;     // 0..1 -> 64 rows
  const int wn   = wave & 3;      // 0..3 -> 32 cols

  const int m_base = blockIdx.y * 128;
  const int n_base = blockIdx.x * 128;

  const int lr   = tid & 127;     // staging row
  const int lseg = tid >> 7;      // 0..1

  const unsigned short* xrow = x   + (long long)(m_base + lr) * 2048;
  const unsigned short* wrow = wbf + (long long)(n_base + lr) * 2048;

  v8f acc[4][2];
  const v8f vzero = {};
#pragma unroll
  for (int i = 0; i < 4; ++i)
#pragma unroll
    for (int j = 0; j < 2; ++j) acc[i][j] = vzero;

  // Prologue: stage K-chunk 0 into buffer 0.
  uint4 xr[2], wr[2];
#pragma unroll
  for (int i = 0; i < 2; ++i) {
    const int seg = lseg * 2 + i;           // 4 segments of 8 halves
    xr[i] = *(const uint4*)&xrow[seg * 8];
    wr[i] = *(const uint4*)&wrow[seg * 8];
  }
#pragma unroll
  for (int i = 0; i < 2; ++i) {
    const int seg = lseg * 2 + i;
    *(uint4*)&lds_x[0][lr * 32 + seg * 8] = xr[i];
    *(uint4*)&lds_w[0][lr * 32 + seg * 8] = wr[i];
  }
  __syncthreads();

  for (int kc = 0; kc < 64; ++kc) {
    const int cur  = kc & 1;
    const bool more = (kc + 1) < 64;

    // Issue next tile's global loads (stay in flight during the WMMAs).
    if (more) {
      const int k0n = (kc + 1) * 32;
#pragma unroll
      for (int i = 0; i < 2; ++i) {
        const int seg = lseg * 2 + i;
        xr[i] = *(const uint4*)&xrow[k0n + seg * 8];
        wr[i] = *(const uint4*)&wrow[k0n + seg * 8];
      }
    }
    if (kc + 8 < 64) {   // long-range L2 prefetch (global_prefetch_b8)
      __builtin_prefetch(xrow + (kc + 8) * 32, 0, 1);
      __builtin_prefetch(wrow + (kc + 8) * 32, 0, 1);
    }

    // Compute from the current buffer.
    v16bf bfr[2];
#pragma unroll
    for (int ct = 0; ct < 2; ++ct) {
      const int n = wn * 32 + ct * 16 + ln;
      bfr[ct] = *(const v16bf*)&lds_w[cur][n * 32 + g * 16];
    }
#pragma unroll
    for (int rt = 0; rt < 4; ++rt) {
      const int row = wm * 64 + rt * 16 + ln;
      v8bf a0 = *(const v8bf*)&lds_x[cur][row * 32 + g * 16];
      v8bf a1 = *(const v8bf*)&lds_x[cur][row * 32 + g * 16 + 8];
      v16bf af = combine16(a0, a1);
#pragma unroll
      for (int ct = 0; ct < 2; ++ct)
        acc[rt][ct] = __builtin_amdgcn_wmma_f32_16x16x32_bf16(
            false, af, false, bfr[ct], (short)0, acc[rt][ct], false, false);
    }

    // Park the next tile into the other buffer; single barrier per iter.
    if (more) {
      const int nxt = cur ^ 1;
#pragma unroll
      for (int i = 0; i < 2; ++i) {
        const int seg = lseg * 2 + i;
        *(uint4*)&lds_x[nxt][lr * 32 + seg * 8] = xr[i];
        *(uint4*)&lds_w[nxt][lr * 32 + seg * 8] = wr[i];
      }
    }
    __syncthreads();
  }

  // Epilogue: add bias, f32 stores.
#pragma unroll
  for (int ct = 0; ct < 2; ++ct) {
    const int n = n_base + wn * 32 + ct * 16 + ln;
    const float bv = bias[n];
#pragma unroll
    for (int rt = 0; rt < 4; ++rt) {
#pragma unroll
      for (int r = 0; r < 8; ++r) {
        const int m = m_base + wm * 64 + rt * 16 + g * 8 + r;
        out[(long long)m * 2048 + n] = acc[rt][ct][r] + bv;
      }
    }
  }
}

extern "C" void kernel_launch(void* const* d_in, const int* in_sizes, int n_in,
                              void* d_out, int out_size, void* d_ws, size_t ws_size,
                              hipStream_t stream) {
  (void)in_sizes; (void)n_in; (void)out_size; (void)ws_size;
  const float* q = (const float*)d_in[0];
  const float* k = (const float*)d_in[1];
  const float* v = (const float*)d_in[2];
  const float* W = (const float*)d_in[3];
  const float* b = (const float*)d_in[4];
  float* out = (float*)d_out;

  // Workspace layout: [0, 32 MiB) X bf16; [32 MiB, 40 MiB) W bf16.
  unsigned short* xws = (unsigned short*)d_ws;
  unsigned short* wbf = xws + (long long)8192 * 2048;

  attn_bf16_wmma_kernel<<<dim3(2048), dim3(128), 0, stream>>>(q, k, v, xws);
  wcvt_kernel<<<dim3(2048), dim3(256), 0, stream>>>(W, wbf);
  proj_bf16_wmma_kernel<<<dim3(16, 64), dim3(256), 0, stream>>>(xws, wbf, b, out);
}
